// ResRNN_34376918237766
// MI455X (gfx1250) — compile-verified, matrix operand order checked by reading, not run on hardware
//
#include <hip/hip_runtime.h>
#include <hip/hip_bf16.h>
#include <math.h>

#define Bdim 64
#define Tdim 2048
#define Ddim 32
#define Hdim 256
#define NWG  16
#define NTHR 128

typedef __bf16 bf16;
typedef __attribute__((ext_vector_type(16))) __bf16 v16bf;
typedef __attribute__((ext_vector_type(8)))  float  v8f;

union BV { v16bf v; uint4 q[2]; };

// Swizzled weight arena: 1 slot = one lane's B-operand image = 16 bf16 = 32B.
// Per weight (K x N), per col-tile ct: (K/32) panels x 32 lane-slots.
#define SL_W1 0        // K=32,  NT=16 ->  512 slots
#define SL_U1 512      // K=256, NT=16 -> 4096
#define SL_W2 4608
#define SL_U2 8704
#define SL_W3 12800
#define SL_U3 16896
#define SL_V3 20992
#define SL_G1 25088    // K=512, NT=16 -> 8192
#define SL_G2 33280    // K=768, NT=16 -> 12288
#define SL_O1 45568    // K=256, NT=2  -> 512
#define SL_O2 46080
#define SL_END 46592   // *32B = 1,490,944 bytes

// ---------------- utility kernels ----------------

__global__ void zero_u32_kernel(unsigned* p, int n) {
  int i = blockIdx.x * blockDim.x + threadIdx.x;
  if (i < n) p[i] = 0u;
}

// Pack fp32 weight W[K][N] (row-major) into per-lane WMMA-B images (bf16).
// Slot (ct, p, lane): lane<16  -> col ct*16+lane,  elems = K{p*32+0..7, p*32+16..23}
//                     lane>=16 -> col ct*16+ln-16, elems = K{p*32+8..15, p*32+24..31}
__global__ void swizzle_kernel(const float* __restrict__ W, bf16* __restrict__ dst,
                               int K, int NT) {
  int idx = blockIdx.x * blockDim.x + threadIdx.x;
  int KP = K >> 5;
  int nslots = NT * KP * 32;
  if (idx >= nslots) return;
  int lane = idx & 31;
  int p    = (idx >> 5) % KP;
  int ct   = (idx >> 5) / KP;
  int N    = NT * 16;
  int c    = ct * 16 + (lane & 15);
  int kh   = lane >> 4;
  bf16* d  = dst + (size_t)idx * 16;
  int kb   = p * 32 + kh * 8;
#pragma unroll
  for (int e = 0; e < 8; e++) d[e]     = (bf16)W[(size_t)(kb + e) * N + c];
#pragma unroll
  for (int e = 0; e < 8; e++) d[8 + e] = (bf16)W[(size_t)(kb + 16 + e) * N + c];
}

// ---------------- device helpers ----------------

__device__ __forceinline__ v8f wmma_bf16(v16bf a, v16bf b, v8f c) {
  return __builtin_amdgcn_wmma_f32_16x16x32_bf16(false, a, false, b, (short)0, c,
                                                 false, false);
}

__device__ __forceinline__ v8f splat8(float x) {
  v8f r;
#pragma unroll
  for (int i = 0; i < 8; i++) r[i] = x;
  return r;
}

__device__ __forceinline__ float sigmoid_fast(float x) {
  return 1.0f / (1.0f + __expf(-x));
}

// tanh(x) = 1 - 2/(exp(2x)+1); exact saturation at +/-inf, one TRANS op.
__device__ __forceinline__ float tanh_fast(float x) {
  return 1.0f - 2.0f / (__expf(2.0f * x) + 1.0f);
}

// A operand (16x32 bf16) from state [64][256] row-major.
// lane<16: K {p*32+0..7, p*32+16..23};  lane>=16: +8 / +24 (kh = lane>>4).
__device__ __forceinline__ v16bf loadA_state(const bf16* s, int m, int p, int kh) {
  const uint4* q = (const uint4*)(s + (size_t)m * Hdim + p * 32 + kh * 8);
  BV u;
  u.q[0] = q[0];
  u.q[1] = q[2];  // +16 elements = +32B
  return u.v;
}

__device__ __forceinline__ v16bf loadB_glob(const uint4* wq, int slot) {
  BV u;
  u.q[0] = wq[2 * slot];
  u.q[1] = wq[2 * slot + 1];
  return u.v;
}

// D layout: comp r -> row mbase+r (lane<16) / mbase+8+r (lane>=16), col = lane&15.
__device__ __forceinline__ void store_tile(bf16* d, v8f v, int mbase, int colbase,
                                           int lane) {
  int c  = colbase + (lane & 15);
  int r0 = mbase + (lane >> 4) * 8;
#pragma unroll
  for (int r = 0; r < 8; r++) d[(size_t)(r0 + r) * Hdim + c] = (bf16)v[r];
}

// Device-scope barrier across the 16 persistent workgroups.
__device__ __forceinline__ void gsync(unsigned* arrive, unsigned* release,
                                      unsigned gen) {
  __threadfence();
  __syncthreads();
  if (threadIdx.x == 0) {
    unsigned a = __hip_atomic_fetch_add(arrive, 1u, __ATOMIC_ACQ_REL,
                                        __HIP_MEMORY_SCOPE_AGENT) + 1u;
    if (a == (unsigned)NWG * gen)
      __hip_atomic_store(release, gen, __ATOMIC_RELEASE, __HIP_MEMORY_SCOPE_AGENT);
    while (__hip_atomic_load(release, __ATOMIC_ACQUIRE, __HIP_MEMORY_SCOPE_AGENT) < gen)
      __builtin_amdgcn_s_sleep(1);
  }
  __syncthreads();
}

// ---------------- persistent RNN kernel ----------------

__global__ __launch_bounds__(NTHR, 1) void rnn_persist(
    const float* __restrict__ x,
    const float* __restrict__ b1, const float* __restrict__ b2,
    const float* __restrict__ b3, const float* __restrict__ bg1,
    const float* __restrict__ bg2, const float* __restrict__ bo,
    const bf16* __restrict__ wswz, bf16* __restrict__ hbuf,
    unsigned* __restrict__ bar, float* __restrict__ out) {
  __shared__ uint4 ldsW[3072];  // U1,W2,U2,W3,U3,V3 column slices, 48 KB

  const int wg   = blockIdx.x;   // 0..15 : owns H columns [wg*16, wg*16+16)
  const int tid  = threadIdx.x;
  const int wid  = tid >> 5;     // wave -> row tile
  const int lane = tid & 31;
  const int kh   = lane >> 4;
  const int lcol = lane & 15;
  const int colbase = wg * 16;
  const int mbase   = wid * 16;
  const int mA      = mbase + lcol;  // batch row this lane loads for A

  const uint4* wq = (const uint4*)wswz;

  // Stage weights into LDS (pre-swizzled in global, simple b128 copies).
  {
    const int slotBase[6] = {SL_U1, SL_W2, SL_U2, SL_W3, SL_U3, SL_V3};
    for (int i = tid; i < 3072; i += NTHR) {
      int widx = i >> 9, off = i & 511;  // 512 uint4 = 256 slots per weight slice
      ldsW[i] = wq[(size_t)(slotBase[widx] + wg * 256) * 2 + off];
    }
  }
  __syncthreads();

  const float vb1 = b1[colbase + lcol], vb2 = b2[colbase + lcol];
  const float vb3 = b3[colbase + lcol];
  const float vg1 = bg1[colbase + lcol], vg2 = bg2[colbase + lcol];

  // Output tile assignment: tile (mt = wg>>1, ct = wg&1) computed by wave wid==mt
  // of WG wg (wg<8) so that its cached h2/h3 A-panels (rows mbase) are reused.
  const bool do_out = (wg < 8) && (wid == (wg >> 1));
  const int  cb     = wg & 1;
  const float vbo   = do_out ? bo[cb * 16 + lcol] : 0.f;

  bf16* h1b = hbuf;
  bf16* h2b = hbuf + 2 * Bdim * Hdim;
  bf16* h3b = hbuf + 4 * Bdim * Hdim;

  // Persistent per-wave register state (elementwise layout == C/D layout).
  v8f h2t = splat8(0.f), h3t = splat8(0.f);
  v8f z1 = splat8(1.f), z2 = splat8(1.f);

  unsigned* arrive  = bar;
  unsigned* release = bar + 1;
  unsigned gen = 0;

  auto ldsB = [&](int widx, int p, int ln) -> v16bf {
    BV u;
    int q = widx * 512 + (p * 32 + ln) * 2;
    u.q[0] = ldsW[q];
    u.q[1] = ldsW[q + 1];
    return u.v;
  };

  for (int t = 0; t < Tdim; t++) {
    const int rb = t & 1, wb = rb ^ 1;
    const bf16* h1c = h1b + rb * Bdim * Hdim;
    bf16*       h1w = h1b + wb * Bdim * Hdim;
    const bf16* h2c = h2b + rb * Bdim * Hdim;
    bf16*       h2w = h2b + wb * Bdim * Hdim;
    const bf16* h3c = h3b + rb * Bdim * Hdim;
    bf16*       h3w = h3b + wb * Bdim * Hdim;

    v8f a0, a1, a2;
    v16bf ac1[8], ac2[8], ac3[8];  // prev-step states (published before this step)
    v16bf ah1[8], ah2[8], ah3[8];  // this-step states (published at barriers)

    // ---- step-start prefetch burst: all old-state A panels + x tile ----
    // h1c/h2c/h3c were all barrier-published last step -> 48 b128 loads in flight,
    // consumed across stages 1-3 (one L2 latency instead of eight per chain).
#pragma unroll
    for (int p = 0; p < 8; p++) ac1[p] = loadA_state(h1c, mA, p, kh);
#pragma unroll
    for (int p = 0; p < 8; p++) ac2[p] = loadA_state(h2c, mA, p, kh);
#pragma unroll
    for (int p = 0; p < 8; p++) ac3[p] = loadA_state(h3c, mA, p, kh);

    BV ax;  // x A-tile, fp32 -> bf16 on the fly (K = D = 32, one panel)
    {
      const float4* px = (const float4*)(x + ((size_t)mA * Tdim + t) * Ddim) + kh * 2;
      float4 f0 = px[0], f1 = px[1], f2 = px[4], f3 = px[5];
      float fx[16] = {f0.x, f0.y, f0.z, f0.w, f1.x, f1.y, f1.z, f1.w,
                      f2.x, f2.y, f2.z, f2.w, f3.x, f3.y, f3.z, f3.w};
#pragma unroll
      for (int e = 0; e < 16; e++) ax.v[e] = (bf16)fx[e];
    }

    // ---- stage 1: h1 = tanh(x_t@W1 + h1@U1 + b1) ----
    a0 = wmma_bf16(ax.v, loadB_glob(wq, SL_W1 + wg * 32 + lane), splat8(vb1));
    a1 = splat8(0.f);
#pragma unroll
    for (int p = 0; p < 8; p += 2) {
      a0 = wmma_bf16(ac1[p], ldsB(0, p, lane), a0);
      a1 = wmma_bf16(ac1[p + 1], ldsB(0, p + 1, lane), a1);
    }
    v8f h1t;
#pragma unroll
    for (int i = 0; i < 8; i++) h1t[i] = tanh_fast(a0[i] + a1[i]);
    store_tile(h1w, h1t, mbase, colbase, lane);
    gsync(arrive, release, ++gen);

    // ---- stage 2: h2 = z1*tanh(h1@W2 + h2@U2 + b2) + (1-z1)*h2 ----
    // batch-load the fresh h1 panels, then run the chain (waits amortized).
#pragma unroll
    for (int p = 0; p < 8; p++) ah1[p] = loadA_state(h1w, mA, p, kh);
    a0 = splat8(vb2);
    a1 = splat8(0.f);
#pragma unroll
    for (int p = 0; p < 8; p++) {
      a0 = wmma_bf16(ah1[p], ldsB(1, p, lane), a0);
      a1 = wmma_bf16(ac2[p], ldsB(2, p, lane), a1);
    }
#pragma unroll
    for (int i = 0; i < 8; i++) {
      float th = tanh_fast(a0[i] + a1[i]);
      h2t[i] = z1[i] * th + (1.f - z1[i]) * h2t[i];
    }
    store_tile(h2w, h2t, mbase, colbase, lane);
    gsync(arrive, release, ++gen);

    // ---- stage 3: h3 = z2*tanh(h2@W3 + h3@U3 + h1@V3 + b3) + (1-z2)*h3 ----
#pragma unroll
    for (int p = 0; p < 8; p++) ah2[p] = loadA_state(h2w, mA, p, kh);
    a0 = splat8(vb3);
    a1 = splat8(0.f);
    a2 = splat8(0.f);
#pragma unroll
    for (int p = 0; p < 8; p++) {
      a0 = wmma_bf16(ah2[p], ldsB(3, p, lane), a0);
      a1 = wmma_bf16(ac3[p], ldsB(4, p, lane), a1);
      a2 = wmma_bf16(ah1[p], ldsB(5, p, lane), a2);
    }
#pragma unroll
    for (int i = 0; i < 8; i++) {
      float th = tanh_fast(a0[i] + a1[i] + a2[i]);
      h3t[i] = z2[i] * th + (1.f - z2[i]) * h3t[i];
    }
    store_tile(h3w, h3t, mbase, colbase, lane);
    gsync(arrive, release, ++gen);

    // ---- stage 4 (local): gates. h3 panel burst issued first, hides under G1.
#pragma unroll
    for (int p = 0; p < 8; p++) ah3[p] = loadA_state(h3w, mA, p, kh);

    a0 = splat8(vg1);
    a1 = splat8(0.f);
#pragma unroll
    for (int p = 0; p < 8; p++) {
      a0 = wmma_bf16(ah1[p], loadB_glob(wq, SL_G1 + wg * 512 + p * 32 + lane), a0);
      a1 = wmma_bf16(ah2[p],
                     loadB_glob(wq, SL_G1 + wg * 512 + (8 + p) * 32 + lane), a1);
    }
    v8f g1v;
#pragma unroll
    for (int i = 0; i < 8; i++) g1v[i] = sigmoid_fast(a0[i] + a1[i]);

    a0 = splat8(vg2);
    a1 = splat8(0.f);
    a2 = splat8(0.f);
#pragma unroll
    for (int p = 0; p < 8; p++) {
      a0 = wmma_bf16(ah1[p], loadB_glob(wq, SL_G2 + wg * 768 + p * 32 + lane), a0);
      a1 = wmma_bf16(ah2[p],
                     loadB_glob(wq, SL_G2 + wg * 768 + (8 + p) * 32 + lane), a1);
      a2 = wmma_bf16(ah3[p],
                     loadB_glob(wq, SL_G2 + wg * 768 + (16 + p) * 32 + lane), a2);
    }
#pragma unroll
    for (int i = 0; i < 8; i++) {
      float s2 = sigmoid_fast(a0[i] + a1[i] + a2[i]);
      z1[i] = g1v[i] * s2;  // z1 <- z1n * z2n
      z2[i] = s2;
    }

    // ---- output: out_t = h2@O1 + h3@O2 + bo (pure WMMA from cached panels) ----
    if (do_out) {
      v8f o0 = splat8(vbo);
      v8f o1 = splat8(0.f);
#pragma unroll
      for (int p = 0; p < 8; p++) {
        o0 = wmma_bf16(ah2[p], loadB_glob(wq, SL_O1 + cb * 256 + p * 32 + lane), o0);
        o1 = wmma_bf16(ah3[p], loadB_glob(wq, SL_O2 + cb * 256 + p * 32 + lane), o1);
      }
      const int dcol = cb * 16 + lcol;
      const int r0   = mbase + kh * 8;
#pragma unroll
      for (int r = 0; r < 8; r++)
        out[((size_t)(r0 + r) * Tdim + t) * Ddim + dcol] = o0[r] + o1[r];
    }
  }
}

// ---------------- host launcher ----------------

extern "C" void kernel_launch(void* const* d_in, const int* in_sizes, int n_in,
                              void* d_out, int out_size, void* d_ws, size_t ws_size,
                              hipStream_t stream) {
  (void)in_sizes; (void)n_in; (void)out_size; (void)ws_size;
  const float* x   = (const float*)d_in[0];
  const float* W1  = (const float*)d_in[1];
  const float* U1  = (const float*)d_in[2];
  const float* b1  = (const float*)d_in[3];
  const float* W2  = (const float*)d_in[4];
  const float* U2  = (const float*)d_in[5];
  const float* b2  = (const float*)d_in[6];
  const float* W3  = (const float*)d_in[7];
  const float* U3  = (const float*)d_in[8];
  const float* V3  = (const float*)d_in[9];
  const float* b3  = (const float*)d_in[10];
  const float* G1  = (const float*)d_in[11];
  const float* bg1 = (const float*)d_in[12];
  const float* G2  = (const float*)d_in[13];
  const float* bg2 = (const float*)d_in[14];
  const float* O1  = (const float*)d_in[15];
  const float* O2  = (const float*)d_in[16];
  const float* bo  = (const float*)d_in[17];
  float* out = (float*)d_out;

  char* ws = (char*)d_ws;
  unsigned* bar = (unsigned*)ws;                              // barrier counters
  bf16* wswz = (bf16*)(ws + 256);                             // swizzled weights
  bf16* hbuf = (bf16*)(ws + 256 + (size_t)SL_END * 32);       // 3 states x 2 bufs

  // Reset barrier + zero initial state every launch (deterministic).
  zero_u32_kernel<<<1, 64, 0, stream>>>(bar, 64);
  int hz = 6 * Bdim * Hdim * 2 / 4;
  zero_u32_kernel<<<(hz + 255) / 256, 256, 0, stream>>>((unsigned*)hbuf, hz);

  auto launch_swz = [&](const float* W, int slotBase, int K, int NT) {
    int nslots = NT * (K >> 5) * 32;
    swizzle_kernel<<<(nslots + 127) / 128, 128, 0, stream>>>(
        W, wswz + (size_t)slotBase * 16, K, NT);
  };
  launch_swz(W1, SL_W1, 32, 16);
  launch_swz(U1, SL_U1, 256, 16);
  launch_swz(W2, SL_W2, 256, 16);
  launch_swz(U2, SL_U2, 256, 16);
  launch_swz(W3, SL_W3, 256, 16);
  launch_swz(U3, SL_U3, 256, 16);
  launch_swz(V3, SL_V3, 256, 16);
  launch_swz(G1, SL_G1, 512, 16);
  launch_swz(G2, SL_G2, 768, 16);
  launch_swz(O1, SL_O1, 256, 2);
  launch_swz(O2, SL_O2, 256, 2);

  rnn_persist<<<NWG, NTHR, 0, stream>>>(x, b1, b2, b3, bg1, bg2, bo, wswz, hbuf,
                                        bar, out);
}
